// ScGNN_25993142075504
// MI455X (gfx1250) — matrix operand order repaired and checked by reading, not compiled, not run on hardware
//
#include <hip/hip_runtime.h>

// ---------------------------------------------------------------------------
// ScGNN bipartite GNN for MI455X (gfx1250, wave32, WMMA + dual-TDM pipeline).
// bf16 GEMM inputs (one-time convert), f32 WMMA accumulate.
// Both A and B tiles are staged by the Tensor Data Mover (wave0/wave1 issue),
// double-buffered in LDS: issue stage k+1, s_wait_tensorcnt 1, compute stage k.
// Edge aggregation uses linearity: segsum(ev*X[s]) @ W == segsum(ev*(X@W)[s]).
// ---------------------------------------------------------------------------

#define N_CELLS 10000
#define N_GENES 4000
#define CELL_F  2000
#define GENE_F  50
#define HID     256
#define LAT     64
#define NCLUST  20
#define N_EDGES 640000

#define CELL_FP 2048   // CELL_F padded to BK multiple
#define GENE_FP 64     // GENE_F padded

typedef __attribute__((ext_vector_type(16))) __bf16 v16bf;
typedef __attribute__((ext_vector_type(8)))  __bf16 v8bf;
typedef __attribute__((ext_vector_type(8)))  float  v8f;
typedef __attribute__((ext_vector_type(4)))  unsigned int u32x4;
typedef __attribute__((ext_vector_type(4)))  int i32x4;
typedef __attribute__((ext_vector_type(8)))  int i32x8;

union ABfrag { v16bf v; v8bf h[2]; };

// float -> bf16 bits, round-to-nearest-even
__device__ __forceinline__ unsigned short f2bf(float f) {
    union { float f; unsigned int u; } x; x.f = f;
    unsigned int r = x.u + 0x7FFFu + ((x.u >> 16) & 1u);
    return (unsigned short)(r >> 16);
}

// ---------------------------------------------------------------------------
// TDM: load a (tile_rows x 64) bf16 tile of a row-major [rows x Kp] tensor
// into LDS with 16B padding after every 128B row (LDS row stride 144B).
// rows_left clamps dim1 so OOB rows read as zero.
// ---------------------------------------------------------------------------
#define BM 128
#define BN 64
#define BK 64
#define AST 72   // LDS row stride in ushorts: 64 data + 8 pad
#define BST 72

#if defined(__gfx1250__)
__device__ __forceinline__ void tdm_load_tile(const unsigned short* gsrc,
                                              unsigned lds_addr,
                                              int rows_left, int Kp,
                                              int tile_rows) {
    unsigned long long ga = (unsigned long long)(uintptr_t)gsrc;
    unsigned td0 = (unsigned)Kp;
    unsigned td1 = (unsigned)rows_left;

    u32x4 g0;
    g0[0] = 1u;                                              // count=1 (valid user D#)
    g0[1] = lds_addr;                                        // LDS byte address
    g0[2] = (unsigned)(ga & 0xFFFFFFFFu);
    g0[3] = (unsigned)((ga >> 32) & 0x01FFFFFFu) | 0x80000000u; // type=2 ("image")

    i32x8 g1;
    // data_size=2B (1<<16), pad_enable (1<<20),
    // pad_interval=32 DWORDs (4<<22), pad_amount=4 DWORDs (3<<25)
    g1[0] = (int)((1u << 16) | (1u << 20) | (4u << 22) | (3u << 25));
    g1[1] = (int)((td0 & 0xFFFFu) << 16);                    // tensor_dim0[15:0]
    g1[2] = (int)((td0 >> 16) | ((td1 & 0xFFFFu) << 16));    // dim0 hi | dim1 lo
    g1[3] = (int)((td1 >> 16) | ((unsigned)BK << 16));       // dim1 hi | tile_dim0=64
    g1[4] = (int)(unsigned)tile_rows;                        // tile_dim1 (tile_dim2=0)
    g1[5] = (int)td0;                                        // tensor_dim0_stride[31:0]
    g1[6] = 0;
    g1[7] = 0;

    i32x4 gz = (i32x4){0, 0, 0, 0};
#if __clang_major__ >= 23
    i32x8 gz8 = (i32x8){0, 0, 0, 0, 0, 0, 0, 0};
    __builtin_amdgcn_tensor_load_to_lds(g0, g1, gz, gz, gz8, 0);
#else
    __builtin_amdgcn_tensor_load_to_lds(g0, g1, gz, gz, 0);
#endif
}
#endif

// ---------------------------------------------------------------------------
// GEMM: C[M,N] = A[M,Kp] @ Bt[N,Kp]^T, bf16 in, f32 accumulate via WMMA.
// Tile BM=128 x BN=64 x BK=64; 8 waves in 4x2; each wave: 32x32 = 2x2 frags.
// Double-buffered TDM staging: wave0 loads A tiles, wave1 loads B tiles.
// Epilogue: + bias1[n] + bias2[n] + ext[m,n], optional relu; dual f32/bf16 out.
// ---------------------------------------------------------------------------
__global__ __launch_bounds__(256)
void gemm_bf16_wmma(const unsigned short* __restrict__ A,
                    const unsigned short* __restrict__ Bt,
                    int M, int N, int Kp,
                    const float* __restrict__ bias1,
                    const float* __restrict__ bias2,
                    const float* __restrict__ ext, int relu,
                    float* __restrict__ Cf, unsigned short* __restrict__ Cb) {
    __shared__ __attribute__((aligned(16))) unsigned short As[2 * BM * AST];
    __shared__ __attribute__((aligned(16))) unsigned short Bs[2 * BN * BST];

    const int tid  = threadIdx.x;
    const int lane = tid & 31;
    const int wave = tid >> 5;
    const int wm   = (wave & 3) * 32;
    const int wn   = (wave >> 2) * 32;
    const int half = lane >> 4;
    const int lm   = lane & 15;

    const int blockN = blockIdx.x * BN;
    const int blockM = blockIdx.y * BM;
    const int nstages = Kp / BK;

    v8f acc[2][2];
    #pragma unroll
    for (int i = 0; i < 2; ++i)
        #pragma unroll
        for (int j = 0; j < 2; ++j) acc[i][j] = (v8f){};

#if defined(__gfx1250__)
    // prologue: stage 0 into buffer 0
    if (wave == 0)
        tdm_load_tile(A + (size_t)blockM * Kp, (unsigned)(uintptr_t)&As[0],
                      M - blockM, Kp, BM);
    else if (wave == 1)
        tdm_load_tile(Bt + (size_t)blockN * Kp, (unsigned)(uintptr_t)&Bs[0],
                      N - blockN, Kp, BN);
#endif

    for (int s = 0; s < nstages; ++s) {
        __syncthreads();   // all waves done computing from buf[(s+1)&1] (stage s-1)

#if defined(__gfx1250__)
        if (wave < 2) {
            if (s + 1 < nstages) {
                int k1 = (s + 1) * BK;
                unsigned buf = (unsigned)((s + 1) & 1);
                if (wave == 0)
                    tdm_load_tile(A + (size_t)blockM * Kp + k1,
                                  (unsigned)(uintptr_t)&As[buf * (BM * AST)],
                                  M - blockM, Kp, BM);
                else
                    tdm_load_tile(Bt + (size_t)blockN * Kp + k1,
                                  (unsigned)(uintptr_t)&Bs[buf * (BN * BST)],
                                  N - blockN, Kp, BN);
                __builtin_amdgcn_s_wait_tensorcnt((short)1);  // stage s complete
            } else {
                __builtin_amdgcn_s_wait_tensorcnt((short)0);  // last stage complete
            }
        }
#endif
        __syncthreads();   // buf[s&1] visible to all waves

        const unsigned short* Ab = &As[(s & 1) * (BM * AST)];
        const unsigned short* Bb = &Bs[(s & 1) * (BN * BST)];

        #pragma unroll
        for (int kk = 0; kk < BK; kk += 32) {
            ABfrag a[2], b[2];
            #pragma unroll
            for (int ti = 0; ti < 2; ++ti) {
                int row = wm + ti * 16 + lm;
                int ka  = kk + (half ? 16 : 0);
                a[ti].h[0] = *(const v8bf*)&Ab[row * AST + ka];
                a[ti].h[1] = *(const v8bf*)&Ab[row * AST + ka + 8];
            }
            #pragma unroll
            for (int tj = 0; tj < 2; ++tj) {
                int col = wn + tj * 16 + lm;
                int kb  = kk + (half ? 16 : 0);
                b[tj].h[0] = *(const v8bf*)&Bb[col * BST + kb];
                b[tj].h[1] = *(const v8bf*)&Bb[col * BST + kb + 8];
            }
            #pragma unroll
            for (int ti = 0; ti < 2; ++ti)
                #pragma unroll
                for (int tj = 0; tj < 2; ++tj)
                    acc[ti][tj] = __builtin_amdgcn_wmma_f32_16x16x32_bf16(
                        false, a[ti].v, false, b[tj].v,
                        (short)0, acc[ti][tj], false, false);
        }
    }

    // ---- epilogue: VGPR r -> M = r + 8*half ; N = lane&15 ----
    #pragma unroll
    for (int ti = 0; ti < 2; ++ti) {
        #pragma unroll
        for (int tj = 0; tj < 2; ++tj) {
            int cn = blockN + wn + tj * 16 + lm;
            if (cn >= N) continue;
            int mbase = blockM + wm + ti * 16 + half * 8;
            #pragma unroll
            for (int r = 0; r < 8; ++r) {
                int cm = mbase + r;
                if (cm >= M) continue;
                float v = acc[ti][tj][r];
                if (bias1) v += bias1[cn];
                if (bias2) v += bias2[cn];
                if (ext)   v += ext[(size_t)cm * N + cn];
                if (relu)  v = fmaxf(v, 0.0f);
                if (Cf) Cf[(size_t)cm * N + cn] = v;
                if (Cb) Cb[(size_t)cm * N + cn] = f2bf(v);
            }
        }
    }
}

// ---------------------------------------------------------------------------
// f32 [M x K] -> bf16 [M x Kp], zero-filled K pad
__global__ void convert_pad(const float* __restrict__ src, unsigned short* __restrict__ dst,
                            int M, int K, int Kp) {
    long long i = (long long)blockIdx.x * blockDim.x + threadIdx.x;
    long long total = (long long)M * Kp;
    if (i >= total) return;
    int m = (int)(i / Kp), k = (int)(i % Kp);
    dst[i] = (k < K) ? f2bf(src[(size_t)m * K + k]) : (unsigned short)0;
}

// f32 W[K x N] -> bf16 Wt[N x Kp] (transposed), zero-filled K pad
__global__ void tconv(const float* __restrict__ W, unsigned short* __restrict__ Wt,
                      int K, int N, int Kp) {
    long long i = (long long)blockIdx.x * blockDim.x + threadIdx.x;
    long long total = (long long)N * Kp;
    if (i >= total) return;
    int n = (int)(i / Kp), k = (int)(i % Kp);
    Wt[i] = (k < K) ? f2bf(W[(size_t)k * N + n]) : (unsigned short)0;
}

// ---------------------------------------------------------------------------
// acc[tgt[e], :HID] += ev[e] * proj[src[e], :HID]; 4 edges/block, float4/thread
__global__ __launch_bounds__(256)
void edge_scatter(const int* __restrict__ tgt, const int* __restrict__ src,
                  const float* __restrict__ ev, const float* __restrict__ proj,
                  float* __restrict__ acc, int nEdges) {
    int sub = threadIdx.x >> 6;
    int f   = (threadIdx.x & 63) * 4;
    int e = blockIdx.x * 4 + sub;
    if (e >= nEdges) return;
    int t = tgt[e], s = src[e];
    float v = ev[e];
    float4 p = *(const float4*)&proj[(size_t)s * HID + f];
    float* a = &acc[(size_t)t * HID + f];
    atomicAdd(a + 0, v * p.x);
    atomicAdd(a + 1, v * p.y);
    atomicAdd(a + 2, v * p.z);
    atomicAdd(a + 3, v * p.w);
}

__global__ void zero_f32(float* __restrict__ p, long long n) {
    long long i = (long long)blockIdx.x * blockDim.x + threadIdx.x;
    if (i < n) p[i] = 0.0f;
}

// ---------------------------------------------------------------------------
// Student-t cluster assignment, ALPHA=1 -> num = 1/(1+dist^2)
__global__ __launch_bounds__(256)
void cluster_q(const float* __restrict__ z, const float* __restrict__ centers,
               float* __restrict__ q) {
    __shared__ float ctr[NCLUST * LAT];
    for (int i = threadIdx.x; i < NCLUST * LAT; i += blockDim.x) ctr[i] = centers[i];
    __syncthreads();
    int cell = blockIdx.x * blockDim.x + threadIdx.x;
    if (cell >= N_CELLS) return;
    float num[NCLUST];
    float s = 0.0f;
    #pragma unroll 1
    for (int c = 0; c < NCLUST; ++c) {
        float d = 0.0f;
        #pragma unroll
        for (int k = 0; k < LAT; ++k) {
            float diff = z[(size_t)cell * LAT + k] - ctr[c * LAT + k];
            d = fmaf(diff, diff, d);
        }
        num[c] = 1.0f / (1.0f + d);
        s += num[c];
    }
    float inv = 1.0f / s;
    for (int c = 0; c < NCLUST; ++c) q[(size_t)cell * NCLUST + c] = num[c] * inv;
}

// ---------------------------------------------------------------------------
static void launch_gemm(const unsigned short* A, const unsigned short* Bt,
                        int M, int N, int Kp,
                        const float* b1, const float* b2, const float* ext,
                        int relu, float* Cf, unsigned short* Cb, hipStream_t s) {
    dim3 grid((N + BN - 1) / BN, (M + BM - 1) / BM);
    gemm_bf16_wmma<<<grid, 256, 0, s>>>(A, Bt, M, N, Kp, b1, b2, ext, relu, Cf, Cb);
}
static void launch_zero(float* p, long long n, hipStream_t s) {
    zero_f32<<<(unsigned)((n + 255) / 256), 256, 0, s>>>(p, n);
}
static void launch_cvt(const float* src, unsigned short* dst, int M, int K, int Kp, hipStream_t s) {
    long long n = (long long)M * Kp;
    convert_pad<<<(unsigned)((n + 255) / 256), 256, 0, s>>>(src, dst, M, K, Kp);
}
static void launch_tconv(const float* W, unsigned short* Wt, int K, int N, int Kp, hipStream_t s) {
    long long n = (long long)N * Kp;
    tconv<<<(unsigned)((n + 255) / 256), 256, 0, s>>>(W, Wt, K, N, Kp);
}

extern "C" void kernel_launch(void* const* d_in, const int* in_sizes, int n_in,
                              void* d_out, int out_size, void* d_ws, size_t ws_size,
                              hipStream_t stream) {
    const float* cell_x  = (const float*)d_in[0];
    const float* gene_x  = (const float*)d_in[1];
    const int*   e_row   = (const int*)  d_in[2];
    const int*   e_col   = (const int*)  d_in[3];
    const float* e_val   = (const float*)d_in[4];
    const float* l0_cs_w = (const float*)d_in[5],  *l0_cs_b = (const float*)d_in[6];
    const float* l0_cn_w = (const float*)d_in[7],  *l0_cn_b = (const float*)d_in[8];
    const float* l0_gs_w = (const float*)d_in[9],  *l0_gs_b = (const float*)d_in[10];
    const float* l0_gn_w = (const float*)d_in[11], *l0_gn_b = (const float*)d_in[12];
    const float* l1_cs_w = (const float*)d_in[13], *l1_cs_b = (const float*)d_in[14];
    const float* l1_cn_w = (const float*)d_in[15], *l1_cn_b = (const float*)d_in[16];
    const float* l1_gs_w = (const float*)d_in[17], *l1_gs_b = (const float*)d_in[18];
    const float* l1_gn_w = (const float*)d_in[19], *l1_gn_b = (const float*)d_in[20];
    const float* cl_w    = (const float*)d_in[21], *cl_b    = (const float*)d_in[22];
    const float* gl_w    = (const float*)d_in[23], *gl_b    = (const float*)d_in[24];
    const float* cd_w    = (const float*)d_in[25], *cd_b    = (const float*)d_in[26];
    const float* gd_w    = (const float*)d_in[27], *gd_b    = (const float*)d_in[28];
    const float* centers = (const float*)d_in[29];

    // ---- workspace carve-up (byte-based; every chunk 16B aligned) ----
    char* p = (char*)d_ws;
    auto take_bf = [&](size_t elems) { unsigned short* r = (unsigned short*)p; p += elems * 2; return r; };
    auto take_f  = [&](size_t elems) { float* r = (float*)p; p += elems * 4; return r; };

    unsigned short* cxb      = take_bf((size_t)N_CELLS * CELL_FP);
    unsigned short* gxb      = take_bf((size_t)N_GENES * GENE_FP);
    unsigned short* wt_l0_cs = take_bf((size_t)HID * CELL_FP);
    unsigned short* wt_l0_cn = take_bf((size_t)HID * GENE_FP);
    unsigned short* wt_l0_gs = take_bf((size_t)HID * GENE_FP);
    unsigned short* wt_l0_gn = take_bf((size_t)HID * CELL_FP);
    unsigned short* wt_l1_cs = take_bf((size_t)HID * HID);
    unsigned short* wt_l1_cn = take_bf((size_t)HID * HID);
    unsigned short* wt_l1_gs = take_bf((size_t)HID * HID);
    unsigned short* wt_l1_gn = take_bf((size_t)HID * HID);
    unsigned short* wt_cl    = take_bf((size_t)LAT * HID);
    unsigned short* wt_gl    = take_bf((size_t)LAT * HID);
    unsigned short* wt_cd    = take_bf((size_t)N_GENES * LAT);
    unsigned short* wt_gd    = take_bf((size_t)N_CELLS * LAT);
    unsigned short* c0b      = take_bf((size_t)N_CELLS * HID);
    unsigned short* g0b      = take_bf((size_t)N_GENES * HID);
    unsigned short* c1b      = take_bf((size_t)N_CELLS * HID);
    unsigned short* g1b      = take_bf((size_t)N_GENES * HID);
    unsigned short* zcb      = take_bf((size_t)N_CELLS * LAT);
    unsigned short* zgb      = take_bf((size_t)N_GENES * LAT);
    float* proj = take_f((size_t)N_CELLS * HID);
    float* accc = take_f((size_t)N_CELLS * HID);
    float* accg = take_f((size_t)N_GENES * HID);

    // ---- output carve-up ----
    float* out      = (float*)d_out;
    float* z_cells  = out;
    float* z_genes  = z_cells + (size_t)N_CELLS * LAT;
    float* cell_rec = z_genes + (size_t)N_GENES * LAT;
    float* gene_rec = cell_rec + (size_t)N_CELLS * N_GENES;
    float* q_out    = gene_rec + (size_t)N_GENES * N_CELLS;

    // ---- one-time bf16 conversion / weight transposition ----
    launch_cvt(cell_x, cxb, N_CELLS, CELL_F, CELL_FP, stream);
    launch_cvt(gene_x, gxb, N_GENES, GENE_F, GENE_FP, stream);
    launch_tconv(l0_cs_w, wt_l0_cs, CELL_F, HID, CELL_FP, stream);
    launch_tconv(l0_cn_w, wt_l0_cn, GENE_F, HID, GENE_FP, stream);
    launch_tconv(l0_gs_w, wt_l0_gs, GENE_F, HID, GENE_FP, stream);
    launch_tconv(l0_gn_w, wt_l0_gn, CELL_F, HID, CELL_FP, stream);
    launch_tconv(l1_cs_w, wt_l1_cs, HID, HID, HID, stream);
    launch_tconv(l1_cn_w, wt_l1_cn, HID, HID, HID, stream);
    launch_tconv(l1_gs_w, wt_l1_gs, HID, HID, HID, stream);
    launch_tconv(l1_gn_w, wt_l1_gn, HID, HID, HID, stream);
    launch_tconv(cl_w, wt_cl, HID, LAT, HID, stream);
    launch_tconv(gl_w, wt_gl, HID, LAT, HID, stream);
    launch_tconv(cd_w, wt_cd, LAT, N_GENES, LAT, stream);
    launch_tconv(gd_w, wt_gd, LAT, N_CELLS, LAT, stream);

    const int SC4 = (N_EDGES + 3) / 4;

    // ---- layer 0 ----
    launch_zero(accc, (long long)N_CELLS * HID, stream);
    launch_gemm(gxb, wt_l0_cn, N_GENES, HID, GENE_FP, nullptr, nullptr, nullptr, 0, proj, nullptr, stream);
    edge_scatter<<<SC4, 256, 0, stream>>>(e_row, e_col, e_val, proj, accc, N_EDGES);
    launch_gemm(cxb, wt_l0_cs, N_CELLS, HID, CELL_FP, l0_cs_b, l0_cn_b, accc, 1, nullptr, c0b, stream);

    launch_zero(accg, (long long)N_GENES * HID, stream);
    launch_gemm(cxb, wt_l0_gn, N_CELLS, HID, CELL_FP, nullptr, nullptr, nullptr, 0, proj, nullptr, stream);
    edge_scatter<<<SC4, 256, 0, stream>>>(e_col, e_row, e_val, proj, accg, N_EDGES);
    launch_gemm(gxb, wt_l0_gs, N_GENES, HID, GENE_FP, l0_gs_b, l0_gn_b, accg, 1, nullptr, g0b, stream);

    // ---- layer 1 (inputs c0b, g0b) ----
    launch_zero(accc, (long long)N_CELLS * HID, stream);
    launch_gemm(g0b, wt_l1_cn, N_GENES, HID, HID, nullptr, nullptr, nullptr, 0, proj, nullptr, stream);
    edge_scatter<<<SC4, 256, 0, stream>>>(e_row, e_col, e_val, proj, accc, N_EDGES);
    launch_gemm(c0b, wt_l1_cs, N_CELLS, HID, HID, l1_cs_b, l1_cn_b, accc, 1, nullptr, c1b, stream);

    launch_zero(accg, (long long)N_GENES * HID, stream);
    launch_gemm(c0b, wt_l1_gn, N_CELLS, HID, HID, nullptr, nullptr, nullptr, 0, proj, nullptr, stream);
    edge_scatter<<<SC4, 256, 0, stream>>>(e_col, e_row, e_val, proj, accg, N_EDGES);
    launch_gemm(g0b, wt_l1_gs, N_GENES, HID, HID, l1_gs_b, l1_gn_b, accg, 1, nullptr, g1b, stream);

    // ---- heads & decoders ----
    launch_gemm(c1b, wt_cl, N_CELLS, LAT, HID, cl_b, nullptr, nullptr, 0, z_cells, zcb, stream);
    launch_gemm(g1b, wt_gl, N_GENES, LAT, HID, gl_b, nullptr, nullptr, 0, z_genes, zgb, stream);
    launch_gemm(zcb, wt_cd, N_CELLS, N_GENES, LAT, cd_b, nullptr, nullptr, 0, cell_rec, nullptr, stream);
    launch_gemm(zgb, wt_gd, N_GENES, N_CELLS, LAT, gd_b, nullptr, nullptr, 0, gene_rec, nullptr, stream);

    // ---- cluster assignment ----
    cluster_q<<<(N_CELLS + 255) / 256, 256, 0, stream>>>(z_cells, centers, q_out);
}
